// ConvAttnPool_33285996544328
// MI455X (gfx1250) — matrix operand order repaired and checked by reading, not compiled, not run on hardware
//
#include <hip/hip_runtime.h>
#include <hip/hip_bf16.h>

// ---- problem constants -----------------------------------------------------
#define VOCAB  51917
#define EDIM   100
#define FDIM   50
#define KW     10
#define YDIM   8921
#define BB     8
#define TT     2500
#define TP     2501     // conv output length (even kernel, pad 5/5)
#define TPAD   2528     // T' padded to multiple of 32
#define TROWS  2544     // embedding rows: T' + KW - 1 = 2536, padded
#define EC     128      // E padded to 128 per kernel-tap slice
#define KIN    1280     // conv GEMM inner dim = KW * EC
#define NKB    40       // KIN / 32
#define FP     64       // F padded to 64
#define YPAD   8960     // Y padded so YTILES divisible by waves/block
#define YTILES 560
#define NWAVE  4        // y-tiles (waves) per attention block
#define NCHUNK 79       // TPAD / 32

typedef __attribute__((ext_vector_type(16))) __bf16 v16bf;
typedef __attribute__((ext_vector_type(8)))  float  v8f;
typedef __attribute__((ext_vector_type(4)))  int    v4i_;

union FragU { v16bf v; uint4 q[2]; };

// A-fragments: two runs of 8 bf16 (bytes +0 / +32). B-fragments: the
// same two runs (bytes +0 / +32). All are 16B-aligned b128 accesses.
__device__ __forceinline__ v16bf ldfrag(const char* p0, const char* p1) {
  FragU f;
  f.q[0] = *(const uint4*)p0;
  f.q[1] = *(const uint4*)p1;
  return f.v;
}

// branch-free tanh: 1 - 2/(exp(2x)+1); exp->inf gives +1, exp->0 gives -1
__device__ __forceinline__ float fast_tanh(float x) {
  float e = __expf(2.f * x);
  return 1.f - 2.f / (e + 1.f);
}

// ---- async global->LDS copy (CDNA5 path) with guarded fallback -------------
#if defined(__has_builtin)
#if __has_builtin(__builtin_amdgcn_global_load_async_to_lds_b128)
#define HAVE_ASYNC_LDS 1
#endif
#endif

__device__ __forceinline__ void fill16(void* lds_dst, const void* gsrc) {
#if defined(HAVE_ASYNC_LDS)
  // builtin signature: (v4i addrspace(1)*, v4i addrspace(3)*, imm offset, imm cpol)
  __builtin_amdgcn_global_load_async_to_lds_b128(
      (__attribute__((address_space(1))) v4i_*)const_cast<void*>(gsrc),
      (__attribute__((address_space(3))) v4i_*)lds_dst, 0, 0);
#else
  *(uint4*)lds_dst = *(const uint4*)gsrc;
#endif
}

__device__ __forceinline__ void fill_wait() {
#if defined(HAVE_ASYNC_LDS)
#if __has_builtin(__builtin_amdgcn_s_wait_asynccnt)
  __builtin_amdgcn_s_wait_asynccnt(0);
#else
  asm volatile("s_wait_asynccnt 0" ::: "memory");
#endif
#endif
}

// ---- prep: embedding gather -> zero-padded bf16 [B][TROWS][128] ------------
__global__ void embed_kernel(const int* __restrict__ x, const float* __restrict__ Wemb,
                             __bf16* __restrict__ Epad) {
  int idx = blockIdx.x * 256 + threadIdx.x;        // B*TROWS*32 threads, 4 ch each
  int c4  = idx & 31;
  int row = (idx >> 5) % TROWS;
  int b   = idx / (32 * TROWS);
  if (b >= BB) return;
  int t = row - 5;                                 // PAD = 5
  float4 w = make_float4(0.f, 0.f, 0.f, 0.f);
  if (t >= 0 && t < TT && c4 < 25) {               // c4==24 covers c=96..99
    int id = x[b * TT + t];
    w = *(const float4*)(Wemb + (size_t)id * EDIM + c4 * 4);
  }
  union { __bf16 v[4]; uint2 u; } pk;
  pk.v[0] = (__bf16)w.x; pk.v[1] = (__bf16)w.y;
  pk.v[2] = (__bf16)w.z; pk.v[3] = (__bf16)w.w;
  *(uint2*)(Epad + ((size_t)(b * TROWS + row) * EC + c4 * 4)) = pk.u;
}

// ---- prep: conv weights -> WmT[f][k*128+c] bf16 ----------------------------
__global__ void wprep_kernel(const float* __restrict__ conv_w, __bf16* __restrict__ WmT) {
  int idx = blockIdx.x * 256 + threadIdx.x;
  if (idx >= FP * KIN) return;
  int f = idx / KIN, ik = idx % KIN;
  int k = ik >> 7, c = ik & 127;
  float v = (f < FDIM && c < EDIM) ? conv_w[(f * EDIM + c) * KW + k] : 0.f;
  WmT[idx] = (__bf16)v;
}

// ---- prep: U_w -> Ubf[y][f] bf16 (padded) ----------------------------------
__global__ void uprep_kernel(const float* __restrict__ Uw, __bf16* __restrict__ Ubf) {
  int idx = blockIdx.x * 256 + threadIdx.x;
  if (idx >= YPAD * FP) return;
  int y = idx >> 6, f = idx & 63;
  float v = (y < YDIM && f < FDIM) ? Uw[y * FDIM + f] : 0.f;
  Ubf[idx] = (__bf16)v;
}

// ---- prep: final_w -> FwT[f][y] f32 (padded) -------------------------------
__global__ void fprep_kernel(const float* __restrict__ Fw, float* __restrict__ FwT) {
  int idx = blockIdx.x * 256 + threadIdx.x;
  if (idx >= FP * YPAD) return;
  int f = idx / YPAD, y = idx % YPAD;
  FwT[idx] = (y < YDIM && f < FDIM) ? Fw[y * FDIM + f] : 0.f;
}

// ---- conv as implicit GEMM, register double-buffered -----------------------
__global__ void __launch_bounds__(32) conv_kernel(
    const __bf16* __restrict__ Epad, const __bf16* __restrict__ WmT,
    const float* __restrict__ conv_b,
    __bf16* __restrict__ h, __bf16* __restrict__ hT) {
  int blk = blockIdx.x;
  int b = blk / 158, mtile = blk % 158;
  int lane = threadIdx.x;
  int m = lane & 15, half = lane >> 4;
  int tp = mtile * 16 + m;                                  // A-row = t'
  const char* Ebase = (const char*)(Epad + (size_t)b * TROWS * EC);

  auto ldA = [&](int kb) {
    int idx0 = kb * 32 + half * 8;                          // A K-runs per lane
    int ks = idx0 >> 7, c0 = idx0 & 127;                    // tap slice, channel
    const char* ar = Ebase + (size_t)(tp + ks) * (EC * 2) + c0 * 2;
    return ldfrag(ar, ar + 32);
  };
  auto ldB = [&](int kb, int nt) {
    const char* wr = (const char*)(WmT + (size_t)(nt * 16 + m) * KIN
                                   + kb * 32 + half * 8);
    return ldfrag(wr, wr + 32);
  };

  v8f acc[4] = {{}, {}, {}, {}};
  v16bf a_cur = ldA(0);
  v16bf b_cur[4];
#pragma unroll
  for (int nt = 0; nt < 4; ++nt) b_cur[nt] = ldB(0, nt);

  for (int kb = 0; kb < NKB; ++kb) {
    int kn = (kb + 1 < NKB) ? kb + 1 : kb;                  // branch-free tail
    v16bf a_nxt = ldA(kn);
    v16bf b_nxt[4];
#pragma unroll
    for (int nt = 0; nt < 4; ++nt) b_nxt[nt] = ldB(kn, nt);
#pragma unroll
    for (int nt = 0; nt < 4; ++nt)
      acc[nt] = __builtin_amdgcn_wmma_f32_16x16x32_bf16(
          false, a_cur, false, b_cur[nt], (short)0, acc[nt], false, false);
    a_cur = a_nxt;
#pragma unroll
    for (int nt = 0; nt < 4; ++nt) b_cur[nt] = b_nxt[nt];
  }

  // epilogue: bias + tanh, transpose through LDS, emit h and hT in bf16
  __shared__ float lds[16 * 65];                            // stride 65: conflict-free
#pragma unroll
  for (int nt = 0; nt < 4; ++nt) {
    int f = nt * 16 + m;                                    // D: lane = f column
    float bias = (f < FDIM) ? conv_b[f] : 0.f;
#pragma unroll
    for (int r = 0; r < 8; ++r) {
      int row = half * 8 + r;                               // D: VGPR r = t' row
      lds[row * 65 + f] = fast_tanh(acc[nt][r] + bias);
    }
  }
  __syncthreads();
  { // h[b][t'][f] row-major: lane -> (row lane/2, f-half)
    int row = lane >> 1, fh = (lane & 1) * 32;
    union { __bf16 v[32]; uint4 q[4]; } pk;
#pragma unroll
    for (int j = 0; j < 32; ++j) pk.v[j] = (__bf16)lds[row * 65 + fh + j];
    char* dst = (char*)(h + ((size_t)b * TPAD + mtile * 16 + row) * FP + fh);
#pragma unroll
    for (int j = 0; j < 4; ++j) *(uint4*)(dst + j * 16) = pk.q[j];
  }
#pragma unroll
  for (int i = 0; i < 2; ++i) { // hT[b][f][t'] for accumulation-GEMM B-frags
    int f = lane + i * 32;
    union { __bf16 v[16]; uint4 q[2]; } pk;
#pragma unroll
    for (int r = 0; r < 16; ++r) pk.v[r] = (__bf16)lds[r * 65 + f];
    char* dst = (char*)(hT + ((size_t)b * FP + f) * TPAD + mtile * 16);
    *(uint4*)dst = pk.q[0];
    *(uint4*)(dst + 16) = pk.q[1];
  }
}

// ---- fused attention: 4 waves/block share LDS-staged h/hT chunks -----------
// scores bounded (|h|<=1, ||U_y||_1 ~ 6) -> exp safe without max subtraction.
__global__ void __launch_bounds__(128) attn_kernel(
    const __bf16* __restrict__ h, const __bf16* __restrict__ hT,
    const __bf16* __restrict__ Ubf, const float* __restrict__ FwT,
    const float* __restrict__ final_b, float* __restrict__ out) {
  int tid = threadIdx.x;
  int lane = tid & 31, wave = tid >> 5;
  int m = lane & 15, half = (lane >> 4) & 1;
  int blk = blockIdx.x;
  int b = blk / (YTILES / NWAVE), grp = blk % (YTILES / NWAVE);
  int yt = grp * NWAVE + wave;                       // this wave's y-tile
  const char* hB  = (const char*)(h  + (size_t)b * TPAD * FP);
  const char* hTB = (const char*)(hT + (size_t)b * FP * TPAD);

  __shared__ __bf16 sh [2][32 * 64];                 // h chunk: [t][f], 4KB each
  __shared__ __bf16 shT[2][64 * 32];                 // hT chunk: [f][t], 4KB each
  __shared__ float red[NWAVE][8][33];

  // cooperative chunk fill: 8KB per chunk, 128 threads x 4 b128 transfers
  auto fill_chunk = [&](int ch, int buf) {
    const char* g0 = hB + (size_t)ch * (32 * FP * 2);        // contiguous 4KB
    char* l0 = (char*)sh[buf];
    fill16(l0 + tid * 16,        g0 + tid * 16);
    fill16(l0 + 2048 + tid * 16, g0 + 2048 + tid * 16);
    int f = tid >> 1, seg = (tid & 1) * 32;                  // hT: 64 rows x 64B
    const char* g1 = hTB + (size_t)f * (TPAD * 2) + (size_t)ch * 64 + seg;
    char* l1 = (char*)shT[buf] + f * 64 + seg;
    fill16(l1, g1);
    fill16(l1 + 16, g1 + 16);
  };

  // per-wave U B-fragments for both f K-steps
  const char* ur = (const char*)(Ubf + (size_t)(yt * 16 + m) * FP);
  v16bf ub[2];
#pragma unroll
  for (int ks = 0; ks < 2; ++ks) {
    const char* p = ur + ks * 64 + half * 16;
    ub[ks] = ldfrag(p, p + 32);
  }

  v8f acc[4] = {{}, {}, {}, {}};        // m[y][f] accumulators (M=y, N=f)
  float lpart = 0.f;                    // per-lane partial softmax denominator

  fill_chunk(0, 0);
  fill_wait();
  __syncthreads();

  for (int ch = 0; ch < NCHUNK; ++ch) {
    int cur = ch & 1, nxt = cur ^ 1;
    if (ch + 1 < NCHUNK) fill_chunk(ch + 1, nxt);   // async prefetch next chunk

    float P0[8], P1[8];
#pragma unroll
    for (int hh = 0; hh < 2; ++hh) {                // two 16-row score tiles
      const char* ar = (const char*)sh[cur] + (size_t)(hh * 16 + m) * (FP * 2);
      v8f s = {};
#pragma unroll
      for (int ks = 0; ks < 2; ++ks) {
        const char* p = ar + ks * 64 + half * 16;
        v16bf a = ldfrag(p, p + 32);
        s = __builtin_amdgcn_wmma_f32_16x16x32_bf16(
            false, a, false, ub[ks], (short)0, s, false, false);
      }
      float* P = hh ? P1 : P0;
#pragma unroll
      for (int r = 0; r < 8; ++r) {                 // S: lane = y col, t row
        int tabs = ch * 32 + hh * 16 + half * 8 + r;
        float e = (tabs < TP) ? __expf(s[r]) : 0.f; // mask padded t'
        P[r] = e;
        lpart += e;
      }
    }
    // S C-frag layout == A-frag layout of the (y x t) GEMM: zero shuffles
    v16bf ap;
#pragma unroll
    for (int i = 0; i < 8; ++i) { ap[i] = (__bf16)P0[i]; ap[8 + i] = (__bf16)P1[i]; }
#pragma unroll
    for (int nt = 0; nt < 4; ++nt) {                // acc[y][f] += P x h
      const char* p = (const char*)shT[cur] + (size_t)((nt * 16 + m) * 32 + half * 8) * 2;
      v16bf bm = ldfrag(p, p + 32);
      acc[nt] = __builtin_amdgcn_wmma_f32_16x16x32_bf16(
          false, ap, false, bm, (short)0, acc[nt], false, false);
    }

    fill_wait();                                    // own async fills done
    __syncthreads();                                // chunk handoff
  }

  // softmax denominator: combine the two lane-halves (same y column)
  float l = lpart + __shfl_xor(lpart, 16, 32);
  // dot with final_w: acc lane = f col, VGPR r = y row (half*8+r)
  float part[8] = {0.f, 0.f, 0.f, 0.f, 0.f, 0.f, 0.f, 0.f};
#pragma unroll
  for (int nt = 0; nt < 4; ++nt) {
    const float* fr = FwT + (size_t)(nt * 16 + m) * YPAD + yt * 16 + half * 8;
    float4 w0 = *(const float4*)fr;
    float4 w1 = *(const float4*)(fr + 4);
    part[0] += acc[nt][0] * w0.x; part[1] += acc[nt][1] * w0.y;
    part[2] += acc[nt][2] * w0.z; part[3] += acc[nt][3] * w0.w;
    part[4] += acc[nt][4] * w1.x; part[5] += acc[nt][5] * w1.y;
    part[6] += acc[nt][6] * w1.z; part[7] += acc[nt][7] * w1.w;
  }
#pragma unroll
  for (int r = 0; r < 8; ++r) red[wave][r][lane] = part[r];
  __syncthreads();
  if (lane < 16) {                       // lane yy holds l for y_local = yy
    int y = yt * 16 + lane;
    if (y < YDIM) {
      int r = lane & 7, lo = (lane < 8) ? 0 : 16;
      float s = 0.f;
#pragma unroll
      for (int j = 0; j < 16; ++j) s += red[wave][r][lo + j];
      out[(size_t)b * YDIM + y] = s / l + final_b[y];
    }
  }
}

// ---- host launch -----------------------------------------------------------
extern "C" void kernel_launch(void* const* d_in, const int* in_sizes, int n_in,
                              void* d_out, int out_size, void* d_ws, size_t ws_size,
                              hipStream_t stream) {
  const int*   x      = (const int*)  d_in[0];
  const float* Wemb   = (const float*)d_in[1];
  const float* conv_w = (const float*)d_in[2];
  const float* conv_b = (const float*)d_in[3];
  const float* U_w    = (const float*)d_in[4];
  const float* fin_w  = (const float*)d_in[5];
  const float* fin_b  = (const float*)d_in[6];
  float* out = (float*)d_out;
  (void)in_sizes; (void)n_in; (void)out_size; (void)ws_size;

  char* ws = (char*)d_ws;
  size_t off = 0;
  auto carve = [&](size_t bytes) {
    char* p = ws + off;
    off += (bytes + 255) & ~(size_t)255;
    return p;
  };
  __bf16* Epad = (__bf16*)carve((size_t)BB * TROWS * EC * 2);   // 5.2 MB
  __bf16* WmT  = (__bf16*)carve((size_t)FP * KIN * 2);          // 160 KB
  __bf16* Ubf  = (__bf16*)carve((size_t)YPAD * FP * 2);         // 1.15 MB
  float*  FwT  = (float*) carve((size_t)FP * YPAD * 4);         // 2.3 MB
  __bf16* hbuf = (__bf16*)carve((size_t)BB * TPAD * FP * 2);    // 2.6 MB
  __bf16* hT   = (__bf16*)carve((size_t)BB * TPAD * FP * 2);    // 2.6 MB

  embed_kernel<<<(BB * TROWS * 32) / 256, 256, 0, stream>>>(x, Wemb, Epad);
  wprep_kernel<<<(FP * KIN + 255) / 256, 256, 0, stream>>>(conv_w, WmT);
  uprep_kernel<<<(YPAD * FP + 255) / 256, 256, 0, stream>>>(U_w, Ubf);
  fprep_kernel<<<(FP * YPAD + 255) / 256, 256, 0, stream>>>(fin_w, FwT);
  conv_kernel<<<BB * 158, 32, 0, stream>>>(Epad, WmT, conv_b, hbuf, hT);
  attn_kernel<<<BB * (YTILES / NWAVE), 128, 0, stream>>>(hbuf, hT, Ubf, FwT, fin_b, out);
}
